// GraphTransformerModelShare_46514495815852
// MI455X (gfx1250) — compile-verified, matrix operand order checked by reading, not run on hardware
//
#include <hip/hip_runtime.h>

// ---------------------------------------------------------------------------
// GraphTransformer forward for MI455X (gfx1250, wave32, WMMA bf16 path)
// B=4, N=1015, K=30, H=256, NH=8, dk=32, L=3, EXTRA=9, M=1024
// ---------------------------------------------------------------------------

typedef __attribute__((ext_vector_type(16))) __bf16 v16bf;
typedef __attribute__((ext_vector_type(8)))  float  v8f;

#define DEV __device__ __forceinline__

constexpr int cB  = 4;
constexpr int cN  = 1015;
constexpr int cKE = 30;
constexpr int cH  = 256;
constexpr int cNH = 8;
constexpr int cM  = 1024;      // N + EXTRA
constexpr int cEX = 9;
constexpr int cNE = cB * cN * cKE;   // 121800 edges
constexpr float cNEG = -1.0e9f;
constexpr float cINVS = 0.17677669529663687f;  // 1/sqrt(32)

DEV unsigned short f2bf(float f) {
  unsigned u = __float_as_uint(f);
  u += 0x7fffu + ((u >> 16) & 1u);          // round-to-nearest-even
  return (unsigned short)(u >> 16);
}
DEV float bf2f(unsigned short h) { return __uint_as_float((unsigned)h << 16); }

union Op32 { uint4 q[2]; v16bf v; };        // one 16x32(A) or 32x16(B) bf16 operand

DEV v8f wmma_bf16(const Op32 a, const Op32 b, v8f c) {
  return __builtin_amdgcn_wmma_f32_16x16x32_bf16(
      false, a.v, false, b.v, (short)0, c, false, false);
}

// ---------------------------------------------------------------------------
// Generic WMMA GEMM:  out = act( A_bf16[nrows,K] @ Wt_bf16[ncols,K]^T + bias
//                                (+ res) )
// One wave computes a 16x32 output tile: two independent accumulators share
// one A operand -> 2 independent v_wmma per k-step (XDL ILP), A traffic /2.
// A buffers are padded to a multiple of 16 rows, so loads are unguarded;
// stores remain row-guarded.
// ---------------------------------------------------------------------------
__global__ __launch_bounds__(128) void gemm_bf16_kernel(
    const unsigned short* __restrict__ A, const unsigned short* __restrict__ Wt,
    const float* __restrict__ bias, const float* __restrict__ res,
    float* __restrict__ outf, unsigned short* __restrict__ outbf,
    int nrows, int ncols, int K, int relu)
{
  const int wave = threadIdx.x >> 5, lane = threadIdx.x & 31;
  const int lo = lane & 15, hi = lane >> 4;
  const int rowTiles = (nrows + 15) >> 4;
  const int colTiles = ncols >> 5;                 // 32-wide column tiles
  const int tile = blockIdx.x * 4 + wave;
  if (tile >= rowTiles * colTiles) return;
  const int tm = tile % rowTiles, tn = tile / rowTiles;

  const unsigned short* Ap  = A  + (size_t)(tm * 16 + lo) * K + hi * 8;
  const unsigned short* Bp0 = Wt + (size_t)(tn * 32 + lo) * K + hi * 8;
  const unsigned short* Bp1 = Bp0 + (size_t)16 * K;

  v8f acc0 = {}, acc1 = {};
  for (int kb = 0; kb < K; kb += 32) {
    Op32 a, b0, b1;
    a.q[0]  = *(const uint4*)(Ap + kb);
    a.q[1]  = *(const uint4*)(Ap + kb + 16);
    b0.q[0] = *(const uint4*)(Bp0 + kb);
    b0.q[1] = *(const uint4*)(Bp0 + kb + 16);
    b1.q[0] = *(const uint4*)(Bp1 + kb);
    b1.q[1] = *(const uint4*)(Bp1 + kb + 16);
    if (kb + 32 < K) {                       // global_prefetch_b8
      __builtin_prefetch(Ap + kb + 32, 0, 1);
      __builtin_prefetch(Bp0 + kb + 32, 0, 1);
      __builtin_prefetch(Bp1 + kb + 32, 0, 1);
    }
    acc0 = wmma_bf16(a, b0, acc0);           // independent accumulators
    acc1 = wmma_bf16(a, b1, acc1);
  }

  const int n0 = tn * 32 + lo, n1 = n0 + 16;
  const float bv0 = bias[n0], bv1 = bias[n1];
#pragma unroll
  for (int r = 0; r < 8; ++r) {
    const int m = tm * 16 + r + hi * 8;
    if (m < nrows) {
      float v0 = acc0[r] + bv0;
      float v1 = acc1[r] + bv1;
      const size_t o0 = (size_t)m * ncols + n0;
      const size_t o1 = (size_t)m * ncols + n1;
      if (res)  { v0 += res[o0]; v1 += res[o1]; }
      if (relu) { v0 = fmaxf(v0, 0.f); v1 = fmaxf(v1, 0.f); }
      if (outf)  { outf[o0]  = v0;       outf[o1]  = v1; }
      if (outbf) { outbf[o0] = f2bf(v0); outbf[o1] = f2bf(v1); }
    }
  }
}

// ---------------------------------------------------------------------------
// Fused attention: one block per (b, h, 16-query tile). 8 waves / block.
// Stage1: S[16,1024] = mask(Q K^T * invs * e + pe) -> bf16 in LDS
// Stage2: row softmax (wave-per-row shuffle reductions)
// Stage3: O = P V via WMMA, cross-wave LDS reduction
// ---------------------------------------------------------------------------
__global__ __launch_bounds__(256) void attn_kernel(
    const unsigned short* __restrict__ Qbf, const unsigned short* __restrict__ Kbf,
    const unsigned short* __restrict__ Vt,  const float* __restrict__ edges,
    const float* __restrict__ hk,
    float* __restrict__ outf, unsigned short* __restrict__ outbf)
{
  __shared__ __align__(16) unsigned short sS[16 * 1024];  // 32KB scores/probs
  __shared__ __align__(16) float sRed[8 * 16 * 32];       // 16KB partial O

  const int bid = blockIdx.x;
  const int qt = bid & 63, h = (bid >> 6) & 7, b = bid >> 9;
  const int wave = threadIdx.x >> 5, lane = threadIdx.x & 31;
  const int lo = lane & 15, hi = lane >> 4;
  const size_t bh = (size_t)(b * cNH + h);

  // ---- stage 1 -------------------------------------------------------------
  const unsigned short* Qp = Qbf + (bh * cM + qt * 16 + lo) * 32;
  Op32 a;
  a.q[0] = *(const uint4*)(Qp + hi * 8);
  a.q[1] = *(const uint4*)(Qp + 16 + hi * 8);

  for (int ct = wave; ct < 64; ct += 8) {
    const unsigned short* Kp = Kbf + (bh * cM + ct * 16 + lo) * 32;
    Op32 kb;
    kb.q[0] = *(const uint4*)(Kp + hi * 8);
    kb.q[1] = *(const uint4*)(Kp + 16 + hi * 8);
    v8f acc = {};
    acc = wmma_bf16(a, kb, acc);

    const int col = ct * 16 + lo;
#pragma unroll
    for (int r = 0; r < 8; ++r) {
      const int ml = r + hi * 8;
      const int row = qt * 16 + ml;
      const float e = edges[((size_t)b * cM + row) * cM + col];
      float s;
      if (e == 0.f) {
        s = cNEG;
      } else {
        const float pe = (row >= cEX && col >= cEX)
            ? hk[((size_t)b * cN + (row - cEX)) * cN + (col - cEX)] : 0.f;
        s = acc[r] * cINVS * e + pe;
      }
      sS[ml * 1024 + col] = f2bf(s);
    }
  }
  __syncthreads();

  // ---- stage 2: softmax ----------------------------------------------------
#pragma unroll
  for (int rr = 0; rr < 2; ++rr) {
    const int row = wave * 2 + rr;
    unsigned short* Sr = sS + row * 1024;
    float mx = -3.4e38f;
    for (int c = lane; c < 1024; c += 32) mx = fmaxf(mx, bf2f(Sr[c]));
    for (int off = 16; off; off >>= 1) mx = fmaxf(mx, __shfl_xor(mx, off, 32));
    float sum = 0.f;
    for (int c = lane; c < 1024; c += 32) {
      const float ex = __expf(bf2f(Sr[c]) - mx);
      sum += ex;
      Sr[c] = f2bf(ex);
    }
    for (int off = 16; off; off >>= 1) sum += __shfl_xor(sum, off, 32);
    const float inv = 1.f / sum;
    for (int c = lane; c < 1024; c += 32) Sr[c] = f2bf(bf2f(Sr[c]) * inv);
  }
  __syncthreads();

  // ---- stage 3: O = P @ V --------------------------------------------------
  v8f a0 = {}, a1 = {};
  const unsigned short* Vp = Vt + bh * 32 * (size_t)cM;
  const int kb0 = wave * 128;
#pragma unroll
  for (int kk = 0; kk < 4; ++kk) {
    const int kb = kb0 + kk * 32;
    Op32 p, v0, v1;
    p.q[0] = *(const uint4*)(sS + lo * 1024 + kb + hi * 8);
    p.q[1] = *(const uint4*)(sS + lo * 1024 + kb + 16 + hi * 8);
    const unsigned short* V0 = Vp + (size_t)lo * cM + kb + hi * 8;
    const unsigned short* V1 = Vp + (size_t)(lo + 16) * cM + kb + hi * 8;
    v0.q[0] = *(const uint4*)(V0); v0.q[1] = *(const uint4*)(V0 + 16);
    v1.q[0] = *(const uint4*)(V1); v1.q[1] = *(const uint4*)(V1 + 16);
    a0 = wmma_bf16(p, v0, a0);
    a1 = wmma_bf16(p, v1, a1);
  }
#pragma unroll
  for (int r = 0; r < 8; ++r) {
    const int ml = r + hi * 8;
    sRed[(wave * 16 + ml) * 32 + lo]      = a0[r];
    sRed[(wave * 16 + ml) * 32 + 16 + lo] = a1[r];
  }
  __syncthreads();

  for (int idx = threadIdx.x; idx < 512; idx += 256) {
    const int ml = idx >> 5, d = idx & 31;
    float s = 0.f;
#pragma unroll
    for (int w2 = 0; w2 < 8; ++w2) s += sRed[(w2 * 16 + ml) * 32 + d];
    const int row = qt * 16 + ml;
    const size_t o = ((size_t)b * cM + row) * cH + h * 32 + d;
    outf[o]  = s;
    outbf[o] = f2bf(s);
  }
}

// ---------------------------------------------------------------------------
// LayerNorm over H=256, one wave per row
// ---------------------------------------------------------------------------
__global__ __launch_bounds__(256) void ln_kernel(
    const float* __restrict__ in, const float* __restrict__ g,
    const float* __restrict__ bt, float* __restrict__ outf,
    unsigned short* __restrict__ outbf, int rows)
{
  const int wave = threadIdx.x >> 5, lane = threadIdx.x & 31;
  const int row = blockIdx.x * 8 + wave;
  if (row >= rows) return;
  const float* x = in + (size_t)row * cH;
  float v[8], s = 0.f;
#pragma unroll
  for (int i = 0; i < 8; ++i) { v[i] = x[lane + 32 * i]; s += v[i]; }
  for (int off = 16; off; off >>= 1) s += __shfl_xor(s, off, 32);
  const float mu = s * (1.f / 256.f);
  float q = 0.f;
#pragma unroll
  for (int i = 0; i < 8; ++i) { const float d = v[i] - mu; q += d * d; }
  for (int off = 16; off; off >>= 1) q += __shfl_xor(q, off, 32);
  const float rstd = rsqrtf(q * (1.f / 256.f) + 1e-5f);
#pragma unroll
  for (int i = 0; i < 8; ++i) {
    const int c = lane + 32 * i;
    const float y = (v[i] - mu) * rstd * g[c] + bt[c];
    outf[(size_t)row * cH + c]  = y;
    outbf[(size_t)row * cH + c] = f2bf(y);
  }
}

// ---------------------------------------------------------------------------
// Small helpers
// ---------------------------------------------------------------------------
__global__ void cvt_bf16_kernel(const float* __restrict__ in,
                                unsigned short* __restrict__ out, int n) {
  const int i = blockIdx.x * blockDim.x + threadIdx.x;
  if (i < n) out[i] = f2bf(in[i]);
}

// W[K,Nout] fp32 -> Wt[Nout,K] bf16 (so WMMA B-operand loads are contiguous)
__global__ void wtrans_kernel(const float* __restrict__ W,
                              unsigned short* __restrict__ Wt, int K, int Nout) {
  const int i = blockIdx.x * blockDim.x + threadIdx.x;
  if (i >= K * Nout) return;
  const int k = i / Nout, n = i % Nout;
  Wt[(size_t)n * K + k] = f2bf(W[i]);
}

// edge weight: w[e] = eh[e,:] . em_W2 + em_b2
__global__ void edge_w_kernel(const unsigned short* __restrict__ eh,
                              const float* __restrict__ w2,
                              const float* __restrict__ b2,
                              float* __restrict__ w, int ne) {
  const int e = blockIdx.x * blockDim.x + threadIdx.x;
  if (e >= ne) return;
  const unsigned short* r = eh + (size_t)e * 128;
  float s = b2[0];
#pragma unroll 16
  for (int j = 0; j < 128; ++j) s += bf2f(r[j]) * w2[j];
  w[e] = s;
}

__global__ void edges_cls_kernel(float* __restrict__ edges) {
  const int i = blockIdx.x * blockDim.x + threadIdx.x;
  if (i >= cB * cN) return;
  const int b = i / cN, j = i % cN;
  const size_t base = (size_t)b * cM * cM;
  edges[base + (j + 1)]              = 1.f;   // edges[b,0,1:N+1]
  edges[base + (size_t)(j + 1) * cM] = 1.f;   // edges[b,1:N+1,0]
}

__global__ void edges_sub_kernel(float* __restrict__ edges,
                                 const int* __restrict__ sub) {
  const int i = blockIdx.x * blockDim.x + threadIdx.x;
  if (i >= cB * cN) return;
  const int b = i / cN, n = i % cN;
  const int sid = sub[i], c = n + cEX;
  const size_t base = (size_t)b * cM * cM;
  edges[base + (size_t)sid * cM + c] = 1.f;
  edges[base + (size_t)c * cM + sid] = 1.f;
}

__global__ void edges_w_scatter_kernel(float* __restrict__ edges,
                                       const int* __restrict__ eidx,
                                       const float* __restrict__ w) {
  const int e = blockIdx.x * blockDim.x + threadIdx.x;
  if (e >= cNE) return;
  const int s = eidx[e], d = eidx[cNE + e];
  const int eb = s / cN;
  const int sl = s % cN + cEX, dl = d % cN + cEX;
  const float wv = w[e];
  const size_t base = (size_t)eb * cM * cM;
  edges[base + (size_t)sl * cM + dl] = wv;
  edges[base + (size_t)dl * cM + sl] = wv;
}

// x = concat(cls_tok, node)  -> fp32 + bf16
__global__ void pack_x_kernel(const float* __restrict__ cls,
                              const float* __restrict__ node,
                              float* __restrict__ xf,
                              unsigned short* __restrict__ xbf) {
  const int i = blockIdx.x * blockDim.x + threadIdx.x;
  if (i >= cB * cM * cH) return;
  const int c = i & 255, m = (i >> 8) & 1023, b = i >> 18;
  const float v = (m < cEX) ? cls[m * cH + c]
                            : node[((size_t)b * cN + (m - cEX)) * cH + c];
  xf[i] = v; xbf[i] = f2bf(v);
}

// split heads: Q,K -> [b,h,m,d] bf16 ; V -> [b,h,d,m] bf16 (transposed)
__global__ void pack_qkv_kernel(const float* __restrict__ qf,
                                const float* __restrict__ kf,
                                const float* __restrict__ vf,
                                unsigned short* __restrict__ qb,
                                unsigned short* __restrict__ kb,
                                unsigned short* __restrict__ vt) {
  const int i = blockIdx.x * blockDim.x + threadIdx.x;
  if (i >= cB * cM * cH) return;
  const int c = i & 255, m = (i >> 8) & 1023, b = i >> 18;
  const int h = c >> 5, d = c & 31;
  const size_t o  = (((size_t)(b * cNH + h)) * cM + m) * 32 + d;
  const size_t ov = (((size_t)(b * cNH + h)) * 32 + d) * cM + m;
  qb[o]  = f2bf(qf[i]);
  kb[o]  = f2bf(kf[i]);
  vt[ov] = f2bf(vf[i]);
}

// ---------------------------------------------------------------------------
// Host driver
// ---------------------------------------------------------------------------
extern "C" void kernel_launch(void* const* d_in, const int* in_sizes, int n_in,
                              void* d_out, int out_size, void* d_ws, size_t ws_size,
                              hipStream_t stream) {
  (void)in_sizes; (void)n_in; (void)out_size; (void)ws_size;

  const float* V      = (const float*)d_in[0];
  const float* E      = (const float*)d_in[1];
  const int*   E_idx  = (const int*)d_in[2];
  const int*   subid  = (const int*)d_in[3];
  const float* hk     = (const float*)d_in[4];
  const float* cls    = (const float*)d_in[6];
  const float* Wq = (const float*)d_in[7],  *bq = (const float*)d_in[8];
  const float* Wk = (const float*)d_in[9],  *bk = (const float*)d_in[10];
  const float* Wv = (const float*)d_in[11], *bv = (const float*)d_in[12];
  const float* Wo = (const float*)d_in[13], *bo = (const float*)d_in[14];
  const float* ln_g = (const float*)d_in[15], *ln_b = (const float*)d_in[16];
  const float* W1 = (const float*)d_in[17], *b1 = (const float*)d_in[18];
  const float* W2 = (const float*)d_in[19], *b2 = (const float*)d_in[20];
  const float* lnf_g = (const float*)d_in[21], *lnf_b = (const float*)d_in[22];
  const float* nmW1 = (const float*)d_in[23], *nmb1 = (const float*)d_in[24];
  const float* nmW2 = (const float*)d_in[25], *nmb2 = (const float*)d_in[26];
  const float* emW1 = (const float*)d_in[27], *emb1 = (const float*)d_in[28];
  const float* emW2 = (const float*)d_in[29], *emb2 = (const float*)d_in[30];

  // workspace bump allocator (256B aligned)
  char* base = (char*)d_ws;
  size_t off = 0;
  auto alloc = [&](size_t bytes) -> void* {
    size_t p = (off + 255) & ~(size_t)255;
    off = p + bytes;
    return (void*)(base + p);
  };

  const int rows_x  = cB * cM;                 // 4096 (multiple of 16)
  const int rows_v  = cB * cN;                 // 4060
  const int rows_vp = (rows_v + 15) & ~15;     // 4064  (padded A rows)
  const int rows_ep = (cNE + 15) & ~15;        // 121808 (padded A rows)

  unsigned short* wq_t  = (unsigned short*)alloc((size_t)cH * cH * 2);
  unsigned short* wk_t  = (unsigned short*)alloc((size_t)cH * cH * 2);
  unsigned short* wv_t  = (unsigned short*)alloc((size_t)cH * cH * 2);
  unsigned short* wo_t  = (unsigned short*)alloc((size_t)cH * cH * 2);
  unsigned short* w1_t  = (unsigned short*)alloc((size_t)cH * 4 * cH * 2);
  unsigned short* w2_t  = (unsigned short*)alloc((size_t)cH * 4 * cH * 2);
  unsigned short* nm1_t = (unsigned short*)alloc((size_t)cH * cH * 2);
  unsigned short* nm2_t = (unsigned short*)alloc((size_t)cH * cH * 2);
  unsigned short* em1_t = (unsigned short*)alloc((size_t)cH * 128 * 2);

  unsigned short* Ebf   = (unsigned short*)alloc((size_t)rows_ep * cH * 2);
  unsigned short* ehbf  = (unsigned short*)alloc((size_t)cNE * 128 * 2);
  float*          wvec  = (float*)alloc((size_t)cNE * 4);
  float*          edges = (float*)alloc((size_t)cB * cM * cM * 4);
  unsigned short* Vbf   = (unsigned short*)alloc((size_t)rows_vp * cH * 2);
  unsigned short* h1bf  = (unsigned short*)alloc((size_t)rows_vp * cH * 2);
  float*          node  = (float*)alloc((size_t)rows_v * cH * 4);
  float*          xf    = (float*)alloc((size_t)rows_x * cH * 4);
  unsigned short* xbf   = (unsigned short*)alloc((size_t)rows_x * cH * 2);
  float* qf = (float*)alloc((size_t)rows_x * cH * 4);
  float* kf = (float*)alloc((size_t)rows_x * cH * 4);
  float* vf = (float*)alloc((size_t)rows_x * cH * 4);
  unsigned short* qbf  = (unsigned short*)alloc((size_t)rows_x * cH * 2);
  unsigned short* kbf  = (unsigned short*)alloc((size_t)rows_x * cH * 2);
  unsigned short* vtbf = (unsigned short*)alloc((size_t)rows_x * cH * 2);
  float*          attnf  = (float*)alloc((size_t)rows_x * cH * 4);
  unsigned short* attnbf = (unsigned short*)alloc((size_t)rows_x * cH * 2);
  float*          projf  = (float*)alloc((size_t)rows_x * cH * 4);
  float*          xaf    = (float*)alloc((size_t)rows_x * cH * 4);
  unsigned short* xabf   = (unsigned short*)alloc((size_t)rows_x * cH * 2);
  unsigned short* hidbf  = (unsigned short*)alloc((size_t)rows_x * 4 * cH * 2);
  float*          ff     = (float*)alloc((size_t)rows_x * cH * 4);

  auto gemm = [&](const unsigned short* A, const unsigned short* Wt,
                  const float* bias, const float* res, float* of,
                  unsigned short* ob, int nr, int nc, int K, int relu) {
    const int tiles = ((nr + 15) / 16) * (nc / 32);
    gemm_bf16_kernel<<<(tiles + 3) / 4, 128, 0, stream>>>(
        A, Wt, bias, res, of, ob, nr, nc, K, relu);
  };
  auto trans = [&](const float* W, unsigned short* Wt, int K, int Nout) {
    wtrans_kernel<<<(K * Nout + 255) / 256, 256, 0, stream>>>(W, Wt, K, Nout);
  };

  // ---- weight prep ---------------------------------------------------------
  trans(Wq, wq_t, cH, cH);   trans(Wk, wk_t, cH, cH);
  trans(Wv, wv_t, cH, cH);   trans(Wo, wo_t, cH, cH);
  trans(W1, w1_t, cH, 4 * cH);
  trans(W2, w2_t, 4 * cH, cH);
  trans(nmW1, nm1_t, cH, cH); trans(nmW2, nm2_t, cH, cH);
  trans(emW1, em1_t, cH, 128);

  cvt_bf16_kernel<<<(cNE * cH + 255) / 256, 256, 0, stream>>>(E, Ebf, cNE * cH);
  cvt_bf16_kernel<<<(rows_v * cH + 255) / 256, 256, 0, stream>>>(V, Vbf, rows_v * cH);

  // ---- edge MLP -> w -------------------------------------------------------
  gemm(Ebf, em1_t, emb1, nullptr, nullptr, ehbf, cNE, 128, cH, 1);
  edge_w_kernel<<<(cNE + 255) / 256, 256, 0, stream>>>(ehbf, emW2, emb2, wvec, cNE);

  // ---- adjacency -----------------------------------------------------------
  hipMemsetAsync(edges, 0, (size_t)cB * cM * cM * 4, stream);
  edges_cls_kernel<<<(cB * cN + 255) / 256, 256, 0, stream>>>(edges);
  edges_sub_kernel<<<(cB * cN + 255) / 256, 256, 0, stream>>>(edges, subid);
  edges_w_scatter_kernel<<<(cNE + 255) / 256, 256, 0, stream>>>(edges, E_idx, wvec);

  // ---- node MLP + token packing -------------------------------------------
  gemm(Vbf, nm1_t, nmb1, nullptr, nullptr, h1bf, rows_v, cH, cH, 1);
  gemm(h1bf, nm2_t, nmb2, nullptr, node, nullptr, rows_v, cH, cH, 1);
  pack_x_kernel<<<(rows_x * cH + 255) / 256, 256, 0, stream>>>(cls, node, xf, xbf);

  // ---- transformer layers --------------------------------------------------
  for (int layer = 0; layer < 3; ++layer) {
    gemm(xbf, wq_t, bq, nullptr, qf, nullptr, rows_x, cH, cH, 0);
    gemm(xbf, wk_t, bk, nullptr, kf, nullptr, rows_x, cH, cH, 0);
    gemm(xbf, wv_t, bv, nullptr, vf, nullptr, rows_x, cH, cH, 0);
    pack_qkv_kernel<<<(rows_x * cH + 255) / 256, 256, 0, stream>>>(
        qf, kf, vf, qbf, kbf, vtbf);

    attn_kernel<<<cB * cNH * (cM / 16), 256, 0, stream>>>(
        qbf, kbf, vtbf, edges, hk, attnf, attnbf);

    gemm(attnbf, wo_t, bo, xf, projf, nullptr, rows_x, cH, cH, 0);
    ln_kernel<<<rows_x / 8, 256, 0, stream>>>(projf, ln_g, ln_b, xaf, xabf, rows_x);

    gemm(xabf, w1_t, b1, nullptr, nullptr, hidbf, rows_x, 4 * cH, cH, 1);
    gemm(hidbf, w2_t, b2, xaf, ff, nullptr, rows_x, cH, 4 * cH, 0);

    float* xdst = (layer == 2) ? (float*)d_out : xf;
    ln_kernel<<<rows_x / 8, 256, 0, stream>>>(ff, lnf_g, lnf_b, xdst, xbf, rows_x);
  }
}
